// WindowAttention_44255343018304
// MI455X (gfx1250) — matrix operand order repaired and checked by reading, not compiled, not run on hardware
//
#include <hip/hip_runtime.h>

typedef __bf16 bf16;
typedef __attribute__((ext_vector_type(16))) __bf16 v16bf;
typedef __attribute__((ext_vector_type(8)))  __bf16 v8bf;
typedef __attribute__((ext_vector_type(4)))  __bf16 v4bf;
typedef __attribute__((ext_vector_type(8)))  float  v8f;

#define B_WIN    4096
#define NTOK     49
#define CDIM     384
#define NH       12
#define HD       32
#define NW       1024
#define QKV_COLS 1152
#define SCALE    0.17677669529663689f

// A-matrix 16x32 bf16 fragment: lane r=lane&15 is row M; half-group h=lane>>4.
// VGPR0-3 hold K = h*8 .. h*8+7, VGPR4-7 hold K = 16+h*8 .. 16+h*8+7.
static __device__ __forceinline__ v16bf load_fragA(const bf16* base, int ld) {
  int lane = threadIdx.x & 31;
  int r = lane & 15;
  int h = lane >> 4;
  const bf16* p = base + r * ld + h * 8;
  v8bf lo = *(const v8bf*)(p);
  v8bf hi = *(const v8bf*)(p + 16);
  v16bf out;
#pragma unroll
  for (int i = 0; i < 8; ++i) { out[i] = lo[i]; out[i + 8] = hi[i]; }
  return out;
}

// B-matrix 32x16 bf16 fragment: lane r=lane&15 is column N; half-group h=lane>>4.
// VGPR0-3 hold K = h*8 .. h*8+7, VGPR4-7 hold K = 16+h*8 .. 16+h*8+7.
static __device__ __forceinline__ v16bf load_fragB(const bf16* base, int ld) {
  int lane = threadIdx.x & 31;
  int r = lane & 15;
  int h = lane >> 4;
  const bf16* p = base + r * ld + h * 8;
  v8bf lo = *(const v8bf*)(p);
  v8bf hi = *(const v8bf*)(p + 16);
  v16bf out;
#pragma unroll
  for (int i = 0; i < 8; ++i) { out[i] = lo[i]; out[i + 8] = hi[i]; }
  return out;
}

static __device__ __forceinline__ v8f wmma_bf(v16bf a, v16bf b, v8f c) {
  return __builtin_amdgcn_wmma_f32_16x16x32_bf16(false, a, false, b, (short)0, c,
                                                 false, false);
}

static __device__ __forceinline__ v4bf cvt4(float4 f) {
  v4bf o;
  o[0] = (bf16)f.x; o[1] = (bf16)f.y; o[2] = (bf16)f.z; o[3] = (bf16)f.w;
  return o;
}

// ---------------------------------------------------------------------------
// Kernel 1: qkv = x @ qkv_w^T + qkv_b  -> q (pre-scaled), k, v, all bf16 in
// identical [b, h, n, d] layout. Block tile 128(M) x 64(N) x 64(K),
// double-buffered LDS, software pipelined. 8 waves, each owns a 16x64 band.
// ---------------------------------------------------------------------------
__global__ __launch_bounds__(256) void qkv_gemm_kernel(
    const float* __restrict__ x, const float* __restrict__ w,
    const float* __restrict__ bias, bf16* __restrict__ qb,
    bf16* __restrict__ kb, bf16* __restrict__ vb) {
  __shared__ __align__(16) bf16 As[2][128 * 72];
  __shared__ __align__(16) bf16 Bs[2][64 * 72];
  const int t = threadIdx.x;
  const int wv = t >> 5;
  const int lane = t & 31;
  const int m0 = blockIdx.y * 128;
  const int n0 = blockIdx.x * 64;
  const float4* __restrict__ xv = (const float4*)x;   // row stride 96 float4
  const float4* __restrict__ wv4 = (const float4*)w;  // row stride 96 float4

  float4 ta[8], tb[4];
  auto load_tiles = [&](int k0) {
    const int kq = k0 >> 2;
#pragma unroll
    for (int i = 0; i < 8; ++i) {
      int ch = t + i * 256;          // 2048 chunks = 128 rows x 16
      int r = ch >> 4, c4 = ch & 15;
      ta[i] = xv[(size_t)(m0 + r) * 96 + kq + c4];
    }
#pragma unroll
    for (int i = 0; i < 4; ++i) {
      int ch = t + i * 256;          // 1024 chunks = 64 rows x 16
      int r = ch >> 4, c4 = ch & 15;
      tb[i] = wv4[(size_t)(n0 + r) * 96 + kq + c4];
    }
  };
  auto store_tiles = [&](int buf) {
#pragma unroll
    for (int i = 0; i < 8; ++i) {
      int ch = t + i * 256;
      int r = ch >> 4, c4 = ch & 15;
      *(v4bf*)&As[buf][r * 72 + c4 * 4] = cvt4(ta[i]);
    }
#pragma unroll
    for (int i = 0; i < 4; ++i) {
      int ch = t + i * 256;
      int r = ch >> 4, c4 = ch & 15;
      *(v4bf*)&Bs[buf][r * 72 + c4 * 4] = cvt4(tb[i]);
    }
  };

  load_tiles(0);
  store_tiles(0);
  __syncthreads();

  v8f acc[4] = {};
  for (int step = 0; step < 6; ++step) {
    const int cur = step & 1;
    if (step < 5) load_tiles((step + 1) * 64);
#pragma unroll
    for (int kk = 0; kk < 64; kk += 32) {
      v16bf a = load_fragA(&As[cur][wv * 16 * 72 + kk], 72);
      v16bf b0 = load_fragB(&Bs[cur][0 * 16 * 72 + kk], 72);
      v16bf b1 = load_fragB(&Bs[cur][1 * 16 * 72 + kk], 72);
      v16bf b2 = load_fragB(&Bs[cur][2 * 16 * 72 + kk], 72);
      v16bf b3 = load_fragB(&Bs[cur][3 * 16 * 72 + kk], 72);
      acc[0] = wmma_bf(a, b0, acc[0]);
      acc[1] = wmma_bf(a, b1, acc[1]);
      acc[2] = wmma_bf(a, b2, acc[2]);
      acc[3] = wmma_bf(a, b3, acc[3]);
    }
    if (step < 5) {
      store_tiles(cur ^ 1);
      __syncthreads();
    }
  }

  const int cL = lane & 15;
  const int rH = lane >> 4;
  int whichA[4], hA[4], dA[4];
  float biasA[4];
#pragma unroll
  for (int nt = 0; nt < 4; ++nt) {
    int c = n0 + nt * 16 + cL;
    whichA[nt] = c / CDIM;
    int rem = c - whichA[nt] * CDIM;
    hA[nt] = rem >> 5;
    dA[nt] = rem & 31;
    biasA[nt] = bias[c];
  }
#pragma unroll
  for (int vv = 0; vv < 8; ++vv) {
    int m = m0 + wv * 16 + vv + 8 * rH;
    int b_ = m / NTOK, n = m - b_ * NTOK;
#pragma unroll
    for (int nt = 0; nt < 4; ++nt) {
      float val = acc[nt][vv] + biasA[nt];
      size_t idx = (((size_t)b_ * NH + hA[nt]) * NTOK + n) * HD + dA[nt];
      if (whichA[nt] == 0)      qb[idx] = (bf16)(val * SCALE);
      else if (whichA[nt] == 1) kb[idx] = (bf16)val;
      else                      vb[idx] = (bf16)val;
    }
  }
}

// ---------------------------------------------------------------------------
// Kernel 2: fused attention per (window, head). 4 waves, N padded 49 -> 64.
// S = q k^T (WMMA); +bias +mask, softmax, lambda-rescale (LDS); O = P v.
// ---------------------------------------------------------------------------
__global__ __launch_bounds__(128) void attn_kernel(
    const bf16* __restrict__ qb, const bf16* __restrict__ kb,
    const bf16* __restrict__ vb, const float* __restrict__ rpb,
    const float* __restrict__ mask, const float* __restrict__ lamb,
    bf16* __restrict__ yb) {
  __shared__ __align__(16) bf16 qs[64 * 40];
  __shared__ __align__(16) bf16 ks[64 * 40];
  __shared__ __align__(16) bf16 vrow[64 * 40];
  __shared__ __align__(16) float ss[64 * 64];
  __shared__ __align__(16) bf16 ps[64 * 72];
  __shared__ __align__(16) bf16 vts[32 * 72];
  const int t = threadIdx.x;
  const int bh = blockIdx.x;
  const int b_ = bh / NH, h = bh - b_ * NH;
  const v8bf* __restrict__ qv = (const v8bf*)(qb + (size_t)bh * NTOK * HD);
  const v8bf* __restrict__ kv = (const v8bf*)(kb + (size_t)bh * NTOK * HD);
  const v8bf* __restrict__ vv8 = (const v8bf*)(vb + (size_t)bh * NTOK * HD);
  v8bf zero8;
#pragma unroll
  for (int i = 0; i < 8; ++i) zero8[i] = (bf16)0.0f;

  // rows of 32 bf16 = 4 chunks of v8bf; 64 rows -> 256 chunks per matrix.
#pragma unroll
  for (int i = 0; i < 2; ++i) {
    int ch = t + i * 128;
    int r = ch >> 2, c8 = ch & 3;
    v8bf q8 = (r < NTOK) ? qv[ch] : zero8;
    v8bf k8 = (r < NTOK) ? kv[ch] : zero8;
    v8bf v8 = (r < NTOK) ? vv8[ch] : zero8;
    *(v8bf*)&qs[r * 40 + c8 * 8] = q8;
    *(v8bf*)&ks[r * 40 + c8 * 8] = k8;
    *(v8bf*)&vrow[r * 40 + c8 * 8] = v8;
  }
  __syncthreads();

  // transpose v (LDS -> LDS) into [d][m] with zero-padded m>=49
  for (int idx = t; idx < 32 * 64; idx += 128) {
    int d = idx >> 6, m = idx & 63;
    vts[d * 72 + m] = vrow[m * 40 + d];
  }

  const int wv = t >> 5;
  const int lane = t & 31;
  const int cL = lane & 15, rH = lane >> 4;
  {
    v16bf a = load_fragA(qs + wv * 16 * 40, 40);
    v16bf b0 = load_fragB(ks + 0 * 16 * 40, 40);
    v16bf b1 = load_fragB(ks + 1 * 16 * 40, 40);
    v16bf b2 = load_fragB(ks + 2 * 16 * 40, 40);
    v16bf b3 = load_fragB(ks + 3 * 16 * 40, 40);
    v8f s0 = {}, s1 = {}, s2 = {}, s3 = {};
    s0 = wmma_bf(a, b0, s0);
    s1 = wmma_bf(a, b1, s1);
    s2 = wmma_bf(a, b2, s2);
    s3 = wmma_bf(a, b3, s3);
#pragma unroll
    for (int vvi = 0; vvi < 8; ++vvi) {
      int rr = (wv * 16 + vvi + 8 * rH) * 64 + cL;
      ss[rr + 0]  = s0[vvi];
      ss[rr + 16] = s1[vvi];
      ss[rr + 32] = s2[vvi];
      ss[rr + 48] = s3[vvi];
    }
  }
  __syncthreads();

  // Row softmax with relative-position bias + window mask + lambda rescale.
  if (t < 64) {
    int n = t;
    bf16* prow = ps + n * 72;
    if (n < NTOK) {
      int i1 = n / 7, j1 = n - i1 * 7;
      const float* mrow = mask + ((size_t)(b_ & (NW - 1)) * NTOK + n) * NTOK;
      float* srow = ss + n * 64;
      float mx = -3.0e38f;
      for (int m = 0; m < NTOK; ++m) {
        int i2 = m / 7, j2 = m - i2 * 7;
        int ridx = (i1 - i2 + 6) * 13 + (j1 - j2 + 6);
        float s = srow[m] + rpb[ridx * NH + h] + mrow[m];
        srow[m] = s;
        mx = fmaxf(mx, s);
      }
      float sum = 0.0f;
      for (int m = 0; m < NTOK; ++m) {
        float e = __expf(srow[m] - mx);
        srow[m] = e;
        sum += e;
      }
      float inv = 1.0f / sum;
      float lam1 = 1.0f + lamb[h];
      const float ad = 1.0f / (float)NTOK;
      for (int m = 0; m < NTOK; ++m) {
        float p = srow[m] * inv;
        prow[m] = (bf16)(ad + (p - ad) * lam1);
      }
      for (int m = NTOK; m < 64; ++m) prow[m] = (bf16)0.0f;
    } else {
      for (int m = 0; m < 64; ++m) prow[m] = (bf16)0.0f;
    }
  }
  __syncthreads();

  v8f oacc[2] = {};
#pragma unroll
  for (int k2 = 0; k2 < 2; ++k2) {
    v16bf a = load_fragA(ps + wv * 16 * 72 + k2 * 32, 72);
    v16bf b0 = load_fragB(vts + 0 * 16 * 72 + k2 * 32, 72);
    v16bf b1 = load_fragB(vts + 1 * 16 * 72 + k2 * 32, 72);
    oacc[0] = wmma_bf(a, b0, oacc[0]);
    oacc[1] = wmma_bf(a, b1, oacc[1]);
  }
#pragma unroll
  for (int vvi = 0; vvi < 8; ++vvi) {
    int n = wv * 16 + vvi + 8 * rH;
    if (n < NTOK) {
      size_t rowb = ((size_t)b_ * NTOK + n) * CDIM + h * HD + cL;
      yb[rowb + 0]  = (bf16)oacc[0][vvi];
      yb[rowb + 16] = (bf16)oacc[1][vvi];
    }
  }
}

// ---------------------------------------------------------------------------
// Kernel 3: out = y @ proj_w^T + proj_b  (y bf16, out fp32),
// same 128x64x64 double-buffered pipeline.
// ---------------------------------------------------------------------------
__global__ __launch_bounds__(256) void proj_gemm_kernel(
    const bf16* __restrict__ y, const float* __restrict__ w,
    const float* __restrict__ bias, float* __restrict__ out) {
  __shared__ __align__(16) bf16 As[2][128 * 72];
  __shared__ __align__(16) bf16 Bs[2][64 * 72];
  const int t = threadIdx.x;
  const int wv = t >> 5;
  const int lane = t & 31;
  const int m0 = blockIdx.y * 128;
  const int n0 = blockIdx.x * 64;
  const v8bf* __restrict__ yv = (const v8bf*)y;       // row stride 48 chunks
  const float4* __restrict__ wv4 = (const float4*)w;  // row stride 96 float4

  v8bf ta[4];
  float4 tb[4];
  auto load_tiles = [&](int k0) {
#pragma unroll
    for (int i = 0; i < 4; ++i) {
      int ch = t + i * 256;          // 1024 chunks = 128 rows x 8 v8bf
      int r = ch >> 3, c8 = ch & 7;
      ta[i] = yv[(size_t)(m0 + r) * 48 + (k0 >> 3) + c8];
    }
#pragma unroll
    for (int i = 0; i < 4; ++i) {
      int ch = t + i * 256;          // 1024 chunks = 64 rows x 16 float4
      int r = ch >> 4, c4 = ch & 15;
      tb[i] = wv4[(size_t)(n0 + r) * 96 + (k0 >> 2) + c4];
    }
  };
  auto store_tiles = [&](int buf) {
#pragma unroll
    for (int i = 0; i < 4; ++i) {
      int ch = t + i * 256;
      int r = ch >> 3, c8 = ch & 7;
      *(v8bf*)&As[buf][r * 72 + c8 * 8] = ta[i];
    }
#pragma unroll
    for (int i = 0; i < 4; ++i) {
      int ch = t + i * 256;
      int r = ch >> 4, c4 = ch & 15;
      *(v4bf*)&Bs[buf][r * 72 + c4 * 4] = cvt4(tb[i]);
    }
  };

  load_tiles(0);
  store_tiles(0);
  __syncthreads();

  v8f acc[4] = {};
  for (int step = 0; step < 6; ++step) {
    const int cur = step & 1;
    if (step < 5) load_tiles((step + 1) * 64);
#pragma unroll
    for (int kk = 0; kk < 64; kk += 32) {
      v16bf a = load_fragA(&As[cur][wv * 16 * 72 + kk], 72);
      v16bf b0 = load_fragB(&Bs[cur][0 * 16 * 72 + kk], 72);
      v16bf b1 = load_fragB(&Bs[cur][1 * 16 * 72 + kk], 72);
      v16bf b2 = load_fragB(&Bs[cur][2 * 16 * 72 + kk], 72);
      v16bf b3 = load_fragB(&Bs[cur][3 * 16 * 72 + kk], 72);
      acc[0] = wmma_bf(a, b0, acc[0]);
      acc[1] = wmma_bf(a, b1, acc[1]);
      acc[2] = wmma_bf(a, b2, acc[2]);
      acc[3] = wmma_bf(a, b3, acc[3]);
    }
    if (step < 5) {
      store_tiles(cur ^ 1);
      __syncthreads();
    }
  }

  const int cL = lane & 15, rH = lane >> 4;
  float biasA[4];
#pragma unroll
  for (int nt = 0; nt < 4; ++nt) biasA[nt] = bias[n0 + nt * 16 + cL];
#pragma unroll
  for (int vv = 0; vv < 8; ++vv) {
    int m = m0 + wv * 16 + vv + 8 * rH;
    size_t rowb = (size_t)m * CDIM + n0 + cL;
#pragma unroll
    for (int nt = 0; nt < 4; ++nt)
      out[rowb + nt * 16] = acc[nt][vv] + biasA[nt];
  }
}

extern "C" void kernel_launch(void* const* d_in, const int* in_sizes, int n_in,
                              void* d_out, int out_size, void* d_ws,
                              size_t ws_size, hipStream_t stream) {
  const float* x      = (const float*)d_in[0];
  const float* mask   = (const float*)d_in[1];
  const float* qkv_w  = (const float*)d_in[2];
  const float* qkv_b  = (const float*)d_in[3];
  const float* proj_w = (const float*)d_in[4];
  const float* proj_b = (const float*)d_in[5];
  const float* rpb    = (const float*)d_in[6];
  const float* lamb   = (const float*)d_in[7];
  float* out = (float*)d_out;

  const size_t elems = (size_t)B_WIN * NH * NTOK * HD;  // 77,070,336
  bf16* qb = (bf16*)d_ws;
  bf16* kb = qb + elems;
  bf16* vb = kb + elems;
  bf16* yb = vb + elems;

  const int Mrows = B_WIN * NTOK;  // 200704, divisible by 128
  qkv_gemm_kernel<<<dim3(QKV_COLS / 64, Mrows / 128), 256, 0, stream>>>(
      x, qkv_w, qkv_b, qb, kb, vb);
  attn_kernel<<<dim3(B_WIN * NH), 128, 0, stream>>>(qb, kb, vb, rpb, mask,
                                                    lamb, yb);
  proj_gemm_kernel<<<dim3(CDIM / 64, Mrows / 128), 256, 0, stream>>>(
      yb, proj_w, proj_b, out);
}
